// Memory_41455024341119
// MI455X (gfx1250) — compile-verified, matrix operand order checked by reading, not run on hardware
//
#include <hip/hip_runtime.h>
#include <hip/hip_bf16.h>

#define ALPHA 0.2f
#define MOM   0.8f
#define NTOK  131072
#define DF    512
#define MMEM  25
#define MPAD  32
#define KAUG  544   // 512 + 32 (padded p columns)

typedef __attribute__((ext_vector_type(16))) __bf16 v16bf;
typedef __attribute__((ext_vector_type(8)))  __bf16 v8bf;
typedef __attribute__((ext_vector_type(8)))  float  v8f;

__device__ __forceinline__ float wred_sum(float v) {
#pragma unroll
    for (int m = 16; m >= 1; m >>= 1) v += __shfl_xor(v, m, 32);
    return v;
}

// Orderable-uint encoding of float for atomicMax (total order).
__device__ __forceinline__ unsigned encf(float f) {
    unsigned b = __float_as_uint(f);
    return (b & 0x80000000u) ? ~b : (b | 0x80000000u);
}
__device__ __forceinline__ float decf(unsigned e) {
    return (e & 0x80000000u) ? __uint_as_float(e ^ 0x80000000u)
                             : __uint_as_float(~e);
}

// Build a 16x32 bf16 WMMA fragment from two contiguous 8-element runs
// (per-lane K runs at +0 and +16 elements; caller folds in 8*h).
__device__ __forceinline__ v16bf ldfrag(const __bf16* ptr) {
    v8bf lo = *(const v8bf*)(ptr);
    v8bf hi = *(const v8bf*)(ptr + 16);
    v16bf f;
#pragma unroll
    for (int i = 0; i < 8; ++i) { f[i] = lo[i]; f[8 + i] = hi[i]; }
    return f;
}

// ---------------- init scratch (colmax/seg/cnt/loss must be zeroed every call) --
__global__ void k_init(unsigned* colmax, float* seg, float* cnt, float* loss) {
    int tid = threadIdx.x;
    if (tid < MPAD) { colmax[tid] = 0u; cnt[tid] = 0.f; }
    if (tid == 0) loss[0] = 0.f;
    for (int i = tid; i < MMEM * DF; i += 256) seg[i] = 0.f;
}

// ---------------- Bbig = [ W1^T ; (cache @ W2^T) pad ] in bf16, layout [o][k] ---
__global__ void k_prep(const float* __restrict__ W, const float* __restrict__ cache,
                       __bf16* __restrict__ Bb) {
    int o = blockIdx.x, tid = threadIdx.x;
    for (int k = tid; k < DF; k += 256) Bb[o * KAUG + k] = (__bf16)W[o * 2 * DF + k];
    if (tid < MMEM) {
        float acc = 0.f;
        for (int j = 0; j < DF; ++j) acc += cache[tid * DF + j] * W[o * 2 * DF + DF + j];
        Bb[o * KAUG + DF + tid] = (__bf16)acc;
    } else if (tid < MPAD) {
        Bb[o * KAUG + DF + tid] = (__bf16)0.0f;
    }
}

// ---------------- p = softmax( normalize(text) @ cache^T ), padded to 32 --------
__global__ void k_scores(const float* __restrict__ text, const float* __restrict__ cache,
                         float* __restrict__ p) {
    __shared__ float cL[MMEM * DF];
    int tid = threadIdx.x;
    for (int i = tid; i < MMEM * DF; i += 256) cL[i] = cache[i];
    __syncthreads();
    int lane = tid & 31, w = tid >> 5;
    int stride = gridDim.x * 8;
    for (int row = blockIdx.x * 8 + w; row < NTOK; row += stride) {
        float tt[16]; float ss = 0.f;
#pragma unroll
        for (int j = 0; j < 16; ++j) {
            tt[j] = text[(size_t)row * DF + lane + 32 * j];
            ss += tt[j] * tt[j];
        }
        ss = wred_sum(ss);
        float rn = 1.0f / fmaxf(sqrtf(ss), 1e-12f);
        float s[MMEM];
#pragma unroll
        for (int m = 0; m < MMEM; ++m) s[m] = 0.f;
#pragma unroll
        for (int j = 0; j < 16; ++j) {
            float t = tt[j]; int k = lane + 32 * j;
#pragma unroll
            for (int m = 0; m < MMEM; ++m) s[m] += t * cL[m * DF + k];
        }
        float mx = -3.0e38f;
#pragma unroll
        for (int m = 0; m < MMEM; ++m) { s[m] = wred_sum(s[m]) * rn; mx = fmaxf(mx, s[m]); }
        float sum = 0.f; float e[MMEM];
#pragma unroll
        for (int m = 0; m < MMEM; ++m) { e[m] = expf(s[m] - mx); sum += e[m]; }
        float inv = 1.0f / sum;
        float pv = 0.f;
#pragma unroll
        for (int m = 0; m < MMEM; ++m) if (lane == m) pv = e[m] * inv;
        p[(size_t)row * MPAD + lane] = pv;   // lanes 25..31 write 0
    }
}

// ---------------- text_fine = ALPHA * ([text|p] @ Bbig) + text  (bf16 WMMA) -----
__global__ __launch_bounds__(256, 1) void k_gemm(const float* __restrict__ text,
                                                 const float* __restrict__ p,
                                                 const __bf16* __restrict__ Bb,
                                                 float* __restrict__ out) {
    __shared__ __attribute__((aligned(16))) __bf16 Al[64 * KAUG];
    int tid = threadIdx.x;
    int blockRow = blockIdx.x * 64;
    for (int idx = tid; idx < 64 * KAUG; idx += 256) {
        int r = idx / KAUG, k = idx - r * KAUG;
        float v = (k < DF) ? text[(size_t)(blockRow + r) * DF + k]
                           : p[(size_t)(blockRow + r) * MPAD + (k - DF)];
        Al[idx] = (__bf16)v;
    }
    __syncthreads();

    int lane = tid & 31, w = tid >> 5;
    int c = lane & 15, h = lane >> 4;
    int rowbase = blockRow + (w & 3) * 16;
    int colbase = blockIdx.y * 256 + (w >> 2) * 128;

    // Per-lane fragment base pointers (fold in the lane's 8*h K-offset so the
    // fully-unrolled loop uses pure immediate offsets).
    const __bf16* Ab = &Al[((w & 3) * 16 + c) * KAUG + 8 * h];
    const __bf16* Bt[8];
#pragma unroll
    for (int t = 0; t < 8; ++t)
        Bt[t] = Bb + (size_t)(colbase + t * 16 + c) * KAUG + 8 * h;

    v8f acc[8];
#pragma unroll
    for (int t = 0; t < 8; ++t)
#pragma unroll
        for (int i = 0; i < 8; ++i) acc[t][i] = 0.0f;

    // Software pipeline: fragments for step kk are loaded during step kk-32.
    v16bf acur = ldfrag(Ab);
    v16bf bcur[8], bnxt[8];
#pragma unroll
    for (int t = 0; t < 8; ++t) bcur[t] = ldfrag(Bt[t]);

#pragma unroll
    for (int kk = 0; kk < KAUG; kk += 32) {
        v16bf anxt;
        if (kk + 32 < KAUG) {
            anxt = ldfrag(Ab + kk + 32);
#pragma unroll
            for (int t = 0; t < 8; ++t) bnxt[t] = ldfrag(Bt[t] + kk + 32);
        }
#pragma unroll
        for (int t = 0; t < 8; ++t) {
            acc[t] = __builtin_amdgcn_wmma_f32_16x16x32_bf16(
                false, acur, false, bcur[t], (short)0, acc[t], false, false);
        }
        acur = anxt;
#pragma unroll
        for (int t = 0; t < 8; ++t) bcur[t] = bnxt[t];
    }

    // C/D layout: row = v + 8h, col = c
#pragma unroll
    for (int t = 0; t < 8; ++t) {
        int col = colbase + t * 16 + c;
#pragma unroll
        for (int v = 0; v < 8; ++v) {
            int row = rowbase + v + 8 * h;
            size_t idx = (size_t)row * DF + col;
            out[idx] = ALPHA * acc[t][v] + text[idx];
        }
    }
}

// ---------------- loss partials: mean |normalize(text_fine) - text| ------------
__global__ void k_loss(const float* __restrict__ out, const float* __restrict__ text,
                       float* __restrict__ loss_g) {
    __shared__ float lacc;
    int tid = threadIdx.x;
    if (tid == 0) lacc = 0.f;
    __syncthreads();
    int lane = tid & 31, w = tid >> 5;
    int stride = gridDim.x * 8;
    for (int row = blockIdx.x * 8 + w; row < NTOK; row += stride) {
        float tf[16]; float ss = 0.f;
#pragma unroll
        for (int j = 0; j < 16; ++j) {
            tf[j] = out[(size_t)row * DF + lane + 32 * j];
            ss += tf[j] * tf[j];
        }
        ss = wred_sum(ss);
        float rn = 1.0f / fmaxf(sqrtf(ss), 1e-12f);
        float part = 0.f;
#pragma unroll
        for (int j = 0; j < 16; ++j)
            part += fabsf(tf[j] * rn - text[(size_t)row * DF + lane + 32 * j]);
        part = wred_sum(part);
        if (lane == 0) atomicAdd(&lacc, part);
    }
    __syncthreads();
    if (tid == 0) atomicAdd(loss_g, lacc);
}

// ---------------- write pass 1: argmax slot + per-column score max -------------
__global__ void k_write1(const float* __restrict__ img, const float* __restrict__ cache,
                         int* __restrict__ assign, float* __restrict__ sa,
                         unsigned* __restrict__ colmax_g) {
    __shared__ float cL[MMEM * DF];
    __shared__ unsigned cm[MPAD];
    int tid = threadIdx.x;
    for (int i = tid; i < MMEM * DF; i += 256) cL[i] = cache[i];
    if (tid < MPAD) cm[tid] = 0u;
    __syncthreads();
    int lane = tid & 31, w = tid >> 5;
    int stride = gridDim.x * 8;
    for (int row = blockIdx.x * 8 + w; row < NTOK; row += stride) {
        float tt[16]; float ss = 0.f;
#pragma unroll
        for (int j = 0; j < 16; ++j) {
            tt[j] = img[(size_t)row * DF + lane + 32 * j];
            ss += tt[j] * tt[j];
        }
        ss = wred_sum(ss);
        float rn = 1.0f / fmaxf(sqrtf(ss), 1e-12f);
        float s[MMEM];
#pragma unroll
        for (int m = 0; m < MMEM; ++m) s[m] = 0.f;
#pragma unroll
        for (int j = 0; j < 16; ++j) {
            float t = tt[j]; int k = lane + 32 * j;
#pragma unroll
            for (int m = 0; m < MMEM; ++m) s[m] += t * cL[m * DF + k];
        }
        float best = -3.0e38f; int bi = 0;
#pragma unroll
        for (int m = 0; m < MMEM; ++m) {
            s[m] = wred_sum(s[m]) * rn;
            if (s[m] > best) { best = s[m]; bi = m; }   // first max, like argmax
        }
        if (lane == 0) { assign[row] = bi; sa[row] = best; }
        float myS = -3.0e38f;
#pragma unroll
        for (int m = 0; m < MMEM; ++m) if (lane == m) myS = s[m];
        if (lane < MMEM) atomicMax(&cm[lane], encf(myS));
    }
    __syncthreads();
    if (tid < MMEM) atomicMax(&colmax_g[tid], cm[tid]);
}

// ---------------- write pass 2: w_i = exp(s_a - colmax[a]); LDS segment sums ---
__global__ void k_write2(const float* __restrict__ img, const int* __restrict__ assign,
                         const float* __restrict__ sa, const unsigned* __restrict__ colmax_g,
                         float* __restrict__ seg_g, float* __restrict__ cnt_g) {
    __shared__ float segL[MMEM * DF];
    __shared__ float cntL[MPAD];
    int tid = threadIdx.x;
    for (int i = tid; i < MMEM * DF; i += 256) segL[i] = 0.f;
    if (tid < MPAD) cntL[tid] = 0.f;
    __syncthreads();
    int lane = tid & 31, w = tid >> 5;
    int stride = gridDim.x * 8;
    for (int row = blockIdx.x * 8 + w; row < NTOK; row += stride) {
        float tt[16]; float ss = 0.f;
#pragma unroll
        for (int j = 0; j < 16; ++j) {
            tt[j] = img[(size_t)row * DF + lane + 32 * j];
            ss += tt[j] * tt[j];
        }
        ss = wred_sum(ss);
        float rn = 1.0f / fmaxf(sqrtf(ss), 1e-12f);
        int a = assign[row];
        float wgt = expf(sa[row] - decf(colmax_g[a])) * rn;
#pragma unroll
        for (int j = 0; j < 16; ++j)
            atomicAdd(&segL[a * DF + lane + 32 * j], wgt * tt[j]);
        if (lane == 0) atomicAdd(&cntL[a], 1.0f);
    }
    __syncthreads();
    for (int i = tid; i < MMEM * DF; i += 256) atomicAdd(&seg_g[i], segL[i]);
    if (tid < MMEM) atomicAdd(&cnt_g[tid], cntL[tid]);
}

// ---------------- finalize: EMA + normalize cache rows; write loss scalar ------
__global__ void k_final(const float* __restrict__ cache, const float* __restrict__ seg_g,
                        const float* __restrict__ cnt_g, const float* __restrict__ loss_g,
                        float* __restrict__ out) {
    int tid = threadIdx.x, lane = tid & 31, w = tid >> 5;
    for (int m = w; m < MMEM; m += 8) {
        float v[16]; float ss = 0.f;
        bool has = cnt_g[m] > 0.f;
#pragma unroll
        for (int j = 0; j < 16; ++j) {
            int k = lane + 32 * j;
            float cv = cache[m * DF + k];
            float nv = has ? (MOM * cv + (1.0f - MOM) * seg_g[m * DF + k]) : cv;
            v[j] = nv; ss += nv * nv;
        }
        ss = wred_sum(ss);
        float rn = 1.0f / fmaxf(sqrtf(ss), 1e-12f);
#pragma unroll
        for (int j = 0; j < 16; ++j)
            out[(size_t)NTOK * DF + 1 + (size_t)m * DF + lane + 32 * j] = v[j] * rn;
    }
    if (tid == 0) out[(size_t)NTOK * DF] = loss_g[0] * (1.0f / ((float)NTOK * (float)DF));
}

extern "C" void kernel_launch(void* const* d_in, const int* in_sizes, int n_in,
                              void* d_out, int out_size, void* d_ws, size_t ws_size,
                              hipStream_t stream) {
    const float* text  = (const float*)d_in[0];
    const float* img   = (const float*)d_in[1];
    const float* W     = (const float*)d_in[2];
    const float* cache = (const float*)d_in[3];
    float* out = (float*)d_out;
    char* ws = (char*)d_ws;

    // workspace layout (all offsets 16B-aligned)
    float*   p      = (float*)ws;                                        // N*32 f32 = 16 MB
    __bf16*  Bb     = (__bf16*)(ws + (size_t)NTOK * MPAD * 4);           // 512*544 bf16
    char*    b2     = ws + (size_t)NTOK * MPAD * 4 + (size_t)DF * KAUG * 2;
    int*     assign = (int*)b2;                                          // N i32
    float*   sa     = (float*)(b2 + (size_t)NTOK * 4);                   // N f32
    unsigned* colmax= (unsigned*)(b2 + (size_t)NTOK * 8);                // 32 u32
    float*   seg    = (float*)(b2 + (size_t)NTOK * 8 + 128);             // 25*512 f32
    float*   cnt    = (float*)(b2 + (size_t)NTOK * 8 + 128 + MMEM * DF * 4);
    float*   lossa  = (float*)(b2 + (size_t)NTOK * 8 + 128 + MMEM * DF * 4 + 128);

    k_init  <<<1, 256, 0, stream>>>(colmax, seg, cnt, lossa);
    k_prep  <<<DF, 256, 0, stream>>>(W, cache, Bb);
    k_scores<<<1024, 256, 0, stream>>>(text, cache, p);
    dim3 gg(NTOK / 64, 2);
    k_gemm  <<<gg, 256, 0, stream>>>(text, p, Bb, out);
    k_loss  <<<1024, 256, 0, stream>>>(out, text, lossa);
    k_write1<<<512, 256, 0, stream>>>(img, cache, assign, sa, colmax);
    k_write2<<<256, 256, 0, stream>>>(img, assign, sa, colmax, seg, cnt);
    k_final <<<1, 256, 0, stream>>>(cache, seg, cnt, lossa, out);
}
